// GINModel_12764642804258
// MI455X (gfx1250) — matrix operand order repaired and checked by reading, not compiled, not run on hardware
//
#include <hip/hip_runtime.h>

typedef __attribute__((ext_vector_type(16))) _Float16 v16h;
typedef __attribute__((ext_vector_type(8)))  float    v8f;

#define TB 256
#define HDIM 256

// ---------------------------------------------------------------- utility
__global__ void k_zero_f32(float* __restrict__ p, long long n) {
  long long i = (long long)blockIdx.x * blockDim.x + threadIdx.x;
  if (i < n) p[i] = 0.0f;
}

__global__ void k_copy_f4(const float4* __restrict__ in, float4* __restrict__ out, long long n4) {
  long long i = (long long)blockIdx.x * blockDim.x + threadIdx.x;
  if (i < n4) out[i] = in[i];
}

// W[K x Hs] f32 row-major  ->  Wt[Hs x K] f16 row-major (transposed)
__global__ void k_w_to_f16t(const float* __restrict__ W, _Float16* __restrict__ Wt, int K, int Hs) {
  int i = blockIdx.x * blockDim.x + threadIdx.x;
  if (i >= K * Hs) return;
  int k = i / Hs, c = i % Hs;
  Wt[(size_t)c * K + k] = (_Float16)W[i];
}

// ---------------------------------------------------------------- scatter-add (GIN aggregation)
template<int F>
__global__ void k_scatter_add(const float* __restrict__ x, float* __restrict__ agg,
                              const long long* __restrict__ srcI,
                              const long long* __restrict__ dstI, long long E) {
  const int F4 = F / 4;
  long long t = (long long)blockIdx.x * blockDim.x + threadIdx.x;
  if (t >= E * (long long)F4) return;
  long long e = t / F4;
  int c = (int)(t % F4);
  long long s = srcI[e], d = dstI[e];
  float4 v = ((const float4*)(x + s * (long long)F))[c];
  float* dp = agg + d * (long long)F + (long long)c * 4;
  unsafeAtomicAdd(dp + 0, v.x);
  unsafeAtomicAdd(dp + 1, v.y);
  unsafeAtomicAdd(dp + 2, v.z);
  unsafeAtomicAdd(dp + 3, v.w);
}

// ---------------------------------------------------------------- WMMA GEMM
// C[N x 256] = act(A[N x K] @ W[K x 256] + bias); W passed transposed f16 (Wt[256 x K]).
// Weights double-buffered through LDS (shared by all 8 waves of the block).
// Optional per-column sum / sum-of-squares (for fused BatchNorm statistics),
// reduced in LDS (ds_add_f32) then flushed once per block.
template<int K>
__global__ __launch_bounds__(256)
void k_gemm_wmma(const float* __restrict__ A, const _Float16* __restrict__ Wt,
                 const float* __restrict__ bias, float* __restrict__ C,
                 int N, int relu, float* __restrict__ stats) {
  constexpr int KSTEPS = K / 32;
  __shared__ __align__(64) _Float16 sW[2][HDIM * 32];  // 2 x 16 KB weight chunks
  __shared__ float sstat[2 * HDIM];

  const int tid  = threadIdx.x;
  const int lane = tid & 31;
  const int wave = tid >> 5;
  const int hi   = lane >> 4;   // half-wave: 0 or 1
  const int lm   = lane & 15;
  const int rowBase = blockIdx.x * 128 + wave * 16;

  sstat[tid] = 0.f;
  sstat[tid + HDIM] = 0.f;

  const v8f vzero = {0.f, 0.f, 0.f, 0.f, 0.f, 0.f, 0.f, 0.f};
  v8f acc[16];
#pragma unroll
  for (int t = 0; t < 16; ++t) acc[t] = vzero;

  // stage weight chunk 0: thread tid owns output column tid (32 halves = 64 B)
  const _Float16* wcol = Wt + (size_t)tid * K;
  {
    float4 w0 = ((const float4*)wcol)[0];
    float4 w1 = ((const float4*)wcol)[1];
    float4* db = (float4*)(&sW[0][tid * 32]);
    db[0] = w0;
    db[1] = w1;
  }
  __syncthreads();

  int arow = rowBase + lm;
  if (arow > N - 1) arow = N - 1;          // clamp: keeps loads valid, EXEC stays full
  const float* ap = A + (size_t)arow * K;

  for (int ks = 0; ks < KSTEPS; ++ks) {
    const int kk = ks * 32;
    const bool more = (ks + 1 < KSTEPS);

    // issue next weight chunk's global loads early (land after the WMMAs)
    const int knext = more ? kk + 32 : 0;
    const float4 nw0 = ((const float4*)(wcol + knext))[0];
    const float4 nw1 = ((const float4*)(wcol + knext))[1];

    // A fragment: lane holds K = {hi*16..+15} (contiguous, matches B)
    const float4 r0 = ((const float4*)(ap + kk + hi * 16))[0];
    const float4 r1 = ((const float4*)(ap + kk + hi * 16))[1];
    const float4 r2 = ((const float4*)(ap + kk + hi * 16))[2];
    const float4 r3 = ((const float4*)(ap + kk + hi * 16))[3];
    v16h a;
    a[0]  = (_Float16)r0.x; a[1]  = (_Float16)r0.y; a[2]  = (_Float16)r0.z; a[3]  = (_Float16)r0.w;
    a[4]  = (_Float16)r1.x; a[5]  = (_Float16)r1.y; a[6]  = (_Float16)r1.z; a[7]  = (_Float16)r1.w;
    a[8]  = (_Float16)r2.x; a[9]  = (_Float16)r2.y; a[10] = (_Float16)r2.z; a[11] = (_Float16)r2.w;
    a[12] = (_Float16)r3.x; a[13] = (_Float16)r3.y; a[14] = (_Float16)r3.z; a[15] = (_Float16)r3.w;

    const _Float16* sbuf = sW[ks & 1];
#pragma unroll
    for (int nt = 0; nt < 16; ++nt) {
      const int col = nt * 16 + lm;        // B layout: N = lane%16, K = 2 halves/VGPR, hi*16 offset
      const v16h b = *(const v16h*)(sbuf + col * 32 + hi * 16);
      acc[nt] = __builtin_amdgcn_wmma_f32_16x16x32_f16(
          false, a, false, b, (short)0, acc[nt], false, false);
    }

    if (more) {
      float4* db = (float4*)(&sW[(ks + 1) & 1][tid * 32]);
      db[0] = nw0;
      db[1] = nw1;
    }
    __syncthreads();
  }

  // epilogue: bias, activation, store, BN partial sums into LDS
#pragma unroll
  for (int nt = 0; nt < 16; ++nt) {
    const int col = nt * 16 + lm;
    const float bb = bias[col];
    float s1 = 0.f, s2 = 0.f;
#pragma unroll
    for (int r = 0; r < 8; ++r) {
      const int row = rowBase + r + 8 * hi;   // C layout: VGPR r -> M = r + 8*hi
      float v = acc[nt][r] + bb;
      if (relu) v = fmaxf(v, 0.f);
      if (row < N) {
        C[(size_t)row * HDIM + col] = v;
        s1 += v;
        s2 += v * v;
      }
    }
    if (stats) {
      atomicAdd(&sstat[col], s1);
      atomicAdd(&sstat[HDIM + col], s2);
    }
  }
  if (stats) {
    __syncthreads();
    unsafeAtomicAdd(stats + tid, sstat[tid]);
    unsafeAtomicAdd(stats + HDIM + tid, sstat[HDIM + tid]);
  }
}

// ---------------------------------------------------------------- BatchNorm(train stats) + ReLU
__global__ void k_bn_relu(const float* __restrict__ hin, float* __restrict__ hout,
                          const float* __restrict__ gamma, const float* __restrict__ beta,
                          const float* __restrict__ stats, long long N) {
  long long i = (long long)blockIdx.x * blockDim.x + threadIdx.x;
  if (i >= N * HDIM) return;
  int c = (int)(i & (HDIM - 1));
  float invN = 1.0f / (float)N;
  float mean = stats[c] * invN;
  float var  = stats[HDIM + c] * invN - mean * mean;
  float inv  = rsqrtf(var + 1e-5f);
  float v = (hin[i] - mean) * inv * gamma[c] + beta[c];
  hout[i] = fmaxf(v, 0.f);
}

// ---------------------------------------------------------------- global mean pool
__global__ void k_pool(const float* __restrict__ h, const long long* __restrict__ bidx,
                       float* __restrict__ sums, float* __restrict__ counts, long long N) {
  const int C4 = HDIM / 4;
  long long t = (long long)blockIdx.x * blockDim.x + threadIdx.x;
  if (t >= N * (long long)C4) return;
  long long node = t / C4;
  int c = (int)(t % C4);
  long long b = bidx[node];
  float4 v = ((const float4*)(h + node * (long long)HDIM))[c];
  float* dp = sums + b * (long long)HDIM + (long long)c * 4;
  unsafeAtomicAdd(dp + 0, v.x);
  unsafeAtomicAdd(dp + 1, v.y);
  unsafeAtomicAdd(dp + 2, v.z);
  unsafeAtomicAdd(dp + 3, v.w);
  if (c == 0) unsafeAtomicAdd(counts + b, 1.0f);
}

__global__ void k_pool_div(const float* __restrict__ sums, const float* __restrict__ counts,
                           float* __restrict__ pooled, int G) {
  int i = blockIdx.x * blockDim.x + threadIdx.x;
  if (i >= G * HDIM) return;
  int g = i >> 8;
  pooled[i] = sums[i] / fmaxf(counts[g], 1.0f);
}

// ---------------------------------------------------------------- predictor tail: 256 -> T
__global__ void k_final(const float* __restrict__ pmid, const float* __restrict__ Wp2,
                        const float* __restrict__ bp2, float* __restrict__ out, int G, int T) {
  int i = blockIdx.x * blockDim.x + threadIdx.x;
  if (i >= G * T) return;
  int g = i / T, t = i % T;
  const float* row = pmid + (size_t)g * HDIM;
  float s = bp2[t];
  for (int k = 0; k < HDIM; ++k) s += row[k] * Wp2[k * T + t];
  out[i] = s;
}

// ---------------------------------------------------------------- host launcher
extern "C" void kernel_launch(void* const* d_in, const int* in_sizes, int n_in,
                              void* d_out, int out_size, void* d_ws, size_t ws_size,
                              hipStream_t stream) {
  const float*     x    = (const float*)d_in[0];
  const long long* eidx = (const long long*)d_in[1];
  const long long* bidx = (const long long*)d_in[2];
  const float* W1_0 = (const float*)d_in[4];
  const float* b1_0 = (const float*)d_in[5];
  const float* W2_0 = (const float*)d_in[6];
  const float* b2_0 = (const float*)d_in[7];
  const float* g0   = (const float*)d_in[8];
  const float* be0  = (const float*)d_in[9];
  const float* W1s  = (const float*)d_in[10];
  const float* b1s  = (const float*)d_in[11];
  const float* W2s  = (const float*)d_in[12];
  const float* b2s  = (const float*)d_in[13];
  const float* gms  = (const float*)d_in[14];
  const float* bts  = (const float*)d_in[15];
  const float* Wp1  = (const float*)d_in[16];
  const float* bp1  = (const float*)d_in[17];
  const float* Wp2  = (const float*)d_in[18];
  const float* bp2  = (const float*)d_in[19];

  const int FIN = 128, H = HDIM;
  const long long N = in_sizes[0] / FIN;
  const long long E = in_sizes[1] / 2;
  const int T   = in_sizes[19];
  const int G   = out_size / T;
  const int Lm1 = in_sizes[11] / H;   // 4 stacked layers

  // workspace carve-up
  char* w = (char*)d_ws;
  size_t off = 0;
  auto carve = [&](size_t bytes) -> void* {
    void* p = w + off;
    off = (off + bytes + 255) & ~(size_t)255;
    return p;
  };
  float* wH    = (float*)carve((size_t)N * H * 4);
  float* wAgg  = (float*)carve((size_t)N * H * 4);
  float* wMid  = (float*)carve((size_t)N * H * 4);
  float* stats = (float*)carve(2 * H * 4);
  float* sums  = (float*)carve((size_t)G * H * 4);
  float* cnts  = (float*)carve((size_t)G * 4);
  float* pooled= (float*)carve((size_t)G * H * 4);
  float* pmid  = (float*)carve((size_t)G * H * 4);
  _Float16* w10t = (_Float16*)carve((size_t)FIN * H * 2);
  _Float16* w20t = (_Float16*)carve((size_t)H * H * 2);
  _Float16* w1st = (_Float16*)carve((size_t)Lm1 * H * H * 2);
  _Float16* w2st = (_Float16*)carve((size_t)Lm1 * H * H * 2);
  _Float16* wp1t = (_Float16*)carve((size_t)H * H * 2);

  auto cdiv = [](long long a, long long b) { return (unsigned)((a + b - 1) / b); };

  // ---- weights -> transposed f16
  k_w_to_f16t<<<cdiv((long long)FIN * H, TB), TB, 0, stream>>>(W1_0, w10t, FIN, H);
  k_w_to_f16t<<<cdiv((long long)H * H, TB), TB, 0, stream>>>(W2_0, w20t, H, H);
  for (int i = 0; i < Lm1; ++i) {
    k_w_to_f16t<<<cdiv((long long)H * H, TB), TB, 0, stream>>>(
        W1s + (size_t)i * H * H, w1st + (size_t)i * H * H, H, H);
    k_w_to_f16t<<<cdiv((long long)H * H, TB), TB, 0, stream>>>(
        W2s + (size_t)i * H * H, w2st + (size_t)i * H * H, H, H);
  }
  k_w_to_f16t<<<cdiv((long long)H * H, TB), TB, 0, stream>>>(Wp1, wp1t, H, H);

  const unsigned gemmBlocks = cdiv(N, 128);
  const long long nElem = N * H;

  // ---- layer 0 (K = 128)
  k_copy_f4<<<cdiv(N * FIN / 4, TB), TB, 0, stream>>>((const float4*)x, (float4*)wAgg, N * FIN / 4);
  k_scatter_add<128><<<cdiv(E * (FIN / 4), TB), TB, 0, stream>>>(x, wAgg, eidx, eidx + E, E);
  k_gemm_wmma<128><<<gemmBlocks, TB, 0, stream>>>(wAgg, w10t, b1_0, wMid, (int)N, 1, nullptr);
  k_zero_f32<<<2, TB, 0, stream>>>(stats, 2 * H);
  k_gemm_wmma<256><<<gemmBlocks, TB, 0, stream>>>(wMid, w20t, b2_0, wAgg, (int)N, 0, stats);
  k_bn_relu<<<cdiv(nElem, TB), TB, 0, stream>>>(wAgg, wH, g0, be0, stats, N);

  // ---- layers 1..Lm1 (K = 256)
  for (int i = 0; i < Lm1; ++i) {
    k_copy_f4<<<cdiv(nElem / 4, TB), TB, 0, stream>>>((const float4*)wH, (float4*)wAgg, nElem / 4);
    k_scatter_add<256><<<cdiv(E * (H / 4), TB), TB, 0, stream>>>(wH, wAgg, eidx, eidx + E, E);
    k_gemm_wmma<256><<<gemmBlocks, TB, 0, stream>>>(
        wAgg, w1st + (size_t)i * H * H, b1s + (size_t)i * H, wMid, (int)N, 1, nullptr);
    k_zero_f32<<<2, TB, 0, stream>>>(stats, 2 * H);
    k_gemm_wmma<256><<<gemmBlocks, TB, 0, stream>>>(
        wMid, w2st + (size_t)i * H * H, b2s + (size_t)i * H, wAgg, (int)N, 0, stats);
    k_bn_relu<<<cdiv(nElem, TB), TB, 0, stream>>>(
        wAgg, wH, gms + (size_t)i * H, bts + (size_t)i * H, stats, N);
  }

  // ---- global mean pool
  k_zero_f32<<<cdiv((long long)G * H, TB), TB, 0, stream>>>(sums, (long long)G * H);
  k_zero_f32<<<cdiv((long long)G, TB), TB, 0, stream>>>(cnts, G);
  k_pool<<<cdiv(N * (H / 4), TB), TB, 0, stream>>>(wH, bidx, sums, cnts, N);
  k_pool_div<<<cdiv((long long)G * H, TB), TB, 0, stream>>>(sums, cnts, pooled, G);

  // ---- predictor
  k_gemm_wmma<256><<<cdiv((long long)G, 128), TB, 0, stream>>>(pooled, wp1t, bp1, pmid, G, 1, nullptr);
  k_final<<<cdiv((long long)G * T, TB), TB, 0, stream>>>(pmid, Wp2, bp2, (float*)d_out, G, T);
}
